// AxialSelfAttention_80169859547955
// MI455X (gfx1250) — matrix operand order repaired and checked by reading, not compiled, not run on hardware
//
#include <hip/hip_runtime.h>
#include <hip/hip_bf16.h>

typedef __attribute__((ext_vector_type(16))) _Float16 v16h;
typedef __attribute__((ext_vector_type(8)))  _Float16 v8h;
typedef __attribute__((ext_vector_type(8)))  float    v8f;
typedef __attribute__((ext_vector_type(4)))  unsigned int v4u;
typedef __attribute__((ext_vector_type(4)))  int v4i;
typedef __attribute__((ext_vector_type(8)))  int v8i;

#define LN_EPS 1e-5f

// Tensor Data Mover availability (device pass only; host pass falls back so the
// host-side parse of kernel bodies stays legal).
#if defined(__has_builtin)
#if __has_builtin(__builtin_amdgcn_tensor_load_to_lds) && \
    __has_builtin(__builtin_amdgcn_s_wait_tensorcnt)
#define USE_TDM 1
#endif
#endif
#ifndef USE_TDM
#define USE_TDM 0
#endif

// ---------------- WMMA helpers (CDNA5 wave32, 16x16x32 f16 -> f32) ----------------

__device__ __forceinline__ v8f wmma_f16(v16h a, v16h b, v8f c) {
  // 8 args: (neg_a, A, neg_b, B, c_mod, C, reuse_a, reuse_b)
  return __builtin_amdgcn_wmma_f32_16x16x32_f16(false, a, false, b, (short)0, c,
                                                false, false);
}

__device__ __forceinline__ v8f v8f_splat(float x) {
  v8f r;
#pragma unroll
  for (int j = 0; j < 8; ++j) r[j] = x;
  return r;
}

// A fragment (16x32 f16, M x K) from a row-major matrix; `base` points at the
// tile origin, `ld` is the row stride in elements. k order used here (contiguous; B packing matches):
//   lanes 0-15  : M=lane,    elems 0-15 = K 0..15
//   lanes 16-31 : M=lane-16, elems 0-15 = K 16..31
__device__ __forceinline__ v16h load_a(const _Float16* base, int ld, int lane) {
  const int m  = lane & 15;
  const int kh = lane >> 4;
  const _Float16* p = base + (size_t)m * ld + kh * 16;
  v8h lo = *(const v8h*)p;
  v8h hi = *(const v8h*)(p + 8);
  v16h a;
#pragma unroll
  for (int i = 0; i < 8; ++i) { a[i] = lo[i]; a[8 + i] = hi[i]; }
  return a;
}

// B fragment (32x16 f16, K x N) from a pre-packed fragment array:
// pk[((frag)*32 + lane)*16 + e]; each lane reads 32 contiguous bytes.
__device__ __forceinline__ v16h load_b(const _Float16* pk, int frag, int lane) {
  const _Float16* p = pk + ((size_t)frag * 32 + lane) * 16;
  v8h lo = *(const v8h*)p;
  v8h hi = *(const v8h*)(p + 8);
  v16h b;
#pragma unroll
  for (int i = 0; i < 8; ++i) { b[i] = lo[i]; b[8 + i] = hi[i]; }
  return b;
}

#if USE_TDM
// Issue a TDM 2D tile load: 16 rows x 768 f16, global row stride T*768 elems,
// into LDS at byte offset lds_off. D# per CDNA5 ISA.
__device__ __forceinline__ void tdm_load_xp_tile(const _Float16* gptr,
                                                 unsigned lds_off) {
  unsigned long long ga = (unsigned long long)(uintptr_t)gptr;
  v4u g0;
  g0[0] = 1u;                                           // count=1, no gather
  g0[1] = lds_off;                                      // lds_addr (bytes)
  g0[2] = (unsigned)(ga & 0xffffffffu);                 // global_addr[95:64]
  g0[3] = (unsigned)((ga >> 32) & 0x01ffffffu) | (2u << 30);  // addr hi | type=2
  v8i g1;
  g1[0] = 0x10000;           // workgroup_mask=0, data_size=1 (2 bytes)
  g1[1] = (int)(768u << 16); // tensor_dim0[15:0] in [31:16] -> 768
  g1[2] = (int)(16u << 16);  // tensor_dim0 hi=0 ; tensor_dim1 lo=16
  g1[3] = (int)(768u << 16); // tensor_dim1 hi=0 ; tile_dim0=768
  g1[4] = 16;                // tile_dim1=16, tile_dim2=0
  g1[5] = 393216;            // tensor_dim0_stride lo32 = 512*768
  g1[6] = 0;                 // stride hi, dim1_stride lo
  g1[7] = 0;
  v4i z = {0, 0, 0, 0};
#if __clang_major__ >= 23
  v8i z8 = {0, 0, 0, 0, 0, 0, 0, 0};
  __builtin_amdgcn_tensor_load_to_lds(g0, g1, z, z, z8, 0);
#else
  __builtin_amdgcn_tensor_load_to_lds(g0, g1, z, z, 0);
#endif
}
#endif

// ---------------- K1: pack weight W (G x Kd, row-major f32) as B = W^T frags --------
// B[k][n] = W[n][k]; tile (kt, nt): lane l holds n = nt*16+(l&15),
// k = kt*32 + (l>>4)*16 + e, e = 0..15 (contiguous along W's row -> coalesced).

__global__ void pack_b_kernel(const float* __restrict__ W, _Float16* __restrict__ pk,
                              int G, int Kd) {
  int tid = blockIdx.x * blockDim.x + threadIdx.x;
  int NT = G >> 4;
  int total = (Kd >> 5) * NT * 32;
  if (tid >= total) return;
  int lane = tid & 31;
  int tile = tid >> 5;
  int kt = tile / NT;
  int nt = tile - kt * NT;
  int n = nt * 16 + (lane & 15);
  int kbase = kt * 32 + (lane >> 4) * 16;
  const float* src = W + (size_t)n * Kd + kbase;
  _Float16* dst = pk + (size_t)tid * 16;
#pragma unroll
  for (int e = 0; e < 16; ++e) dst[e] = (_Float16)src[e];
}

// ---------------- K2: per-(b,t) tile transpose + LayerNorm + f16 convert ------------
// inputs (B,N,T,K) f32 -> xraw (B,T,K,N) f16 and xln (B,T,K,N) f16 (LN over N).

__global__ void ln_convert_kernel(const float* __restrict__ x,
                                  const float* __restrict__ lnw,
                                  const float* __restrict__ lnb,
                                  _Float16* __restrict__ xraw,
                                  _Float16* __restrict__ xln) {
  const int N = 128, T = 512, K = 64;
  __shared__ float tile[128 * 64];
  __shared__ float redm[4][64];
  __shared__ float redv[4][64];
  __shared__ float meanv[64];
  __shared__ float invv[64];
  __shared__ float wsh[128];
  __shared__ float bsh[128];
  int bt = blockIdx.x;               // b*T + t
  int b = bt >> 9, t = bt & 511;
  int tid = threadIdx.x;             // 256 threads
  if (tid < 128) { wsh[tid] = lnw[tid]; bsh[tid] = lnb[tid]; }
  const float* src = x + (size_t)b * N * T * K + (size_t)t * K;
  for (int i = tid; i < N * K; i += 256) {
    int n = i >> 6, k = i & 63;
    tile[i] = src[(size_t)n * T * K + k];
  }
  __syncthreads();
  {
    int k = tid & 63, part = tid >> 6;
    float s = 0.f, ss = 0.f;
    for (int n = part * 32; n < part * 32 + 32; ++n) {
      float v = tile[n * 64 + k];
      s += v; ss += v * v;
    }
    redm[part][k] = s; redv[part][k] = ss;
  }
  __syncthreads();
  if (tid < 64) {
    float s  = redm[0][tid] + redm[1][tid] + redm[2][tid] + redm[3][tid];
    float ss = redv[0][tid] + redv[1][tid] + redv[2][tid] + redv[3][tid];
    float mu = s * (1.f / 128.f);
    float var = ss * (1.f / 128.f) - mu * mu;
    meanv[tid] = mu;
    invv[tid] = rsqrtf(var + LN_EPS);
  }
  __syncthreads();
  _Float16* orow = xraw + (size_t)bt * K * N;
  _Float16* lrow = xln  + (size_t)bt * K * N;
  for (int i = tid; i < N * K; i += 256) {
    int k = i >> 7, n = i & 127;
    float v = tile[n * 64 + k];
    orow[(size_t)k * N + n] = (_Float16)v;
    lrow[(size_t)k * N + n] = (_Float16)((v - meanv[k]) * invv[k] * wsh[n] + bsh[n]);
  }
}

// ---------------- K3: xp = xln @ w_ih^T + b_ih  (M=262144, K=128, N=768) ------------
// xp stored as (s, t, 768) f16 with s = b*64 + kbin, row r = s*512 + t.

__global__ void __launch_bounds__(256) gemm_xp_kernel(
    const _Float16* __restrict__ xln, const _Float16* __restrict__ wpk,
    const float* __restrict__ b_ih, _Float16* __restrict__ xp) {
  const int T = 512, K = 64, N = 128, G = 768;
  int wave = threadIdx.x >> 5, lane = threadIdx.x & 31;
  int mt = blockIdx.x * 8 + wave;            // 16384 M-tiles
  int r0 = mt * 16;
  int s = r0 >> 9, t0 = r0 & 511;
  int b = s >> 6, kk = s & 63;
  const _Float16* abase = xln + ((size_t)(b * T + t0) * K + kk) * N;
  const int lda = K * N;                     // consecutive t -> stride K*N
  v16h afr[4];
#pragma unroll
  for (int kt = 0; kt < 4; ++kt) afr[kt] = load_a(abase + kt * 32, lda, lane);
  _Float16* orow = xp + (size_t)r0 * G;
  for (int nt = 0; nt < 48; ++nt) {
    v8f acc = v8f_splat(b_ih[nt * 16 + (lane & 15)]);
#pragma unroll
    for (int kt = 0; kt < 4; ++kt)
      acc = wmma_f16(afr[kt], load_b(wpk, kt * 48 + nt, lane), acc);
    int n = nt * 16 + (lane & 15);
#pragma unroll
    for (int j = 0; j < 8; ++j) {
      int m = j + 8 * (lane >> 4);
      orow[(size_t)m * G + n] = (_Float16)acc[j];
    }
  }
}

// ---------------- K4: GRU recurrence, 16 sequences / workgroup, h in LDS ------------
// Per step: 16x256 @ 256x768 via 384 WMMAs, gates elementwise. The per-step xp
// slice (16x768 f16 = 24KB) is double-buffered in LDS by the Tensor Data Mover,
// issued by wave 0 so the DMA overlaps the gate GEMM; s_wait_tensorcnt + the
// workgroup barrier publish it before the gate phase.

__global__ void __launch_bounds__(256, 1) gru_kernel(
    const _Float16* __restrict__ xp, const _Float16* __restrict__ whh,
    const float* __restrict__ b_hh, _Float16* __restrict__ hs) {
  const int T = 512, H = 256, G = 768;
  __shared__ _Float16 hbuf[16 * 256];
  __shared__ float gbuf[16 * 768];
#if USE_TDM
  __shared__ _Float16 xpbuf[2][16 * 768];
#endif
  int tid = threadIdx.x, wave = tid >> 5, lane = tid & 31;
  int s0 = blockIdx.x * 16;
  for (int i = tid; i < 16 * 256; i += 256) hbuf[i] = (_Float16)0.f;
#if USE_TDM
  // generic LDS pointers carry the LDS byte offset in their low 32 bits
  unsigned xpb_off[2];
  xpb_off[0] = (unsigned)(uintptr_t)(&xpbuf[0][0]);
  xpb_off[1] = (unsigned)(uintptr_t)(&xpbuf[1][0]);
  if (wave == 0)
    tdm_load_xp_tile(xp + (size_t)s0 * T * G, xpb_off[0]);  // t = 0
#endif
  __syncthreads();
  for (int t = 0; t < T; ++t) {
#if USE_TDM
    if (wave == 0 && t + 1 < T)
      tdm_load_xp_tile(xp + ((size_t)s0 * T + (t + 1)) * G, xpb_off[(t + 1) & 1]);
#endif
    // phase A: g = h @ w_hh^T + b_hh  (16x256 @ 256x768): 8 waves x 6 N-tiles x 8 K
#pragma unroll 1
    for (int ntl = 0; ntl < 6; ++ntl) {
      int nt = wave * 6 + ntl;
      v8f acc = v8f_splat(b_hh[nt * 16 + (lane & 15)]);
#pragma unroll
      for (int kt = 0; kt < 8; ++kt)
        acc = wmma_f16(load_a(&hbuf[kt * 32], 256, lane),
                       load_b(whh, kt * 48 + nt, lane), acc);
      int n = nt * 16 + (lane & 15);
#pragma unroll
      for (int j = 0; j < 8; ++j) {
        int m = j + 8 * (lane >> 4);
        gbuf[m * 768 + n] = acc[j];
      }
    }
#if USE_TDM
    if (wave == 0) {
      // tensor ops complete in order: <=1 outstanding means tile t has landed
      if (t + 1 < T) __builtin_amdgcn_s_wait_tensorcnt(1);
      else           __builtin_amdgcn_s_wait_tensorcnt(0);
    }
#endif
    __syncthreads();
    // phase B: gate math, 4096 h elements, 16 per thread
#pragma unroll 1
    for (int it = 0; it < 16; ++it) {
      int idx = it * 256 + tid;
      int i = idx >> 8, j = idx & 255;
#if USE_TDM
      const _Float16* xrow = &xpbuf[t & 1][i * 768];
      float xr = (float)xrow[j];
      float xz = (float)xrow[256 + j];
      float xn = (float)xrow[512 + j];
#else
      size_t ro = ((size_t)(s0 + i) * T + t) * G;
      float xr = (float)xp[ro + j];
      float xz = (float)xp[ro + 256 + j];
      float xn = (float)xp[ro + 512 + j];
      if (t + 1 < T) __builtin_prefetch(&xp[ro + G + j], 0, 1);
#endif
      float hr = gbuf[i * 768 + j];
      float hz = gbuf[i * 768 + 256 + j];
      float hn = gbuf[i * 768 + 512 + j];
      float rr = 1.f / (1.f + __expf(-(xr + hr)));
      float zz = 1.f / (1.f + __expf(-(xz + hz)));
      float nn = tanhf(xn + rr * hn);
      float hp = (float)hbuf[i * 256 + j];
      float hv = (1.f - zz) * nn + zz * hp;
      hbuf[i * 256 + j] = (_Float16)hv;
      hs[((size_t)(s0 + i) * T + t) * H + j] = (_Float16)hv;
    }
    __syncthreads();
  }
}

// ---------------- K5: out_t = hs @ w_fc^T + b_fc, stored as packed B frags ----------
// Per (b,t): B-matrix for att@out_t is (K=64 bins x C=64), 8 frags * 512 f16 = 4096.

__global__ void __launch_bounds__(256) gemm_outt_kernel(
    const _Float16* __restrict__ hs, const _Float16* __restrict__ wfc,
    const float* __restrict__ b_fc, _Float16* __restrict__ opk) {
  const int T = 512, H = 256;
  int wave = threadIdx.x >> 5, lane = threadIdx.x & 31;
  int mt = blockIdx.x * 8 + wave;
  int r0 = mt * 16;
  int s = r0 >> 9, t0 = r0 & 511;
  int b = s >> 6, kbin = s & 63;
  const _Float16* abase = hs + (size_t)r0 * H;
  v16h afr[8];
#pragma unroll
  for (int kt = 0; kt < 8; ++kt) afr[kt] = load_a(abase + kt * 32, H, lane);
  int kt2 = kbin >> 5;
  int lh  = (kbin >> 4) & 1;
  int e   = kbin & 15;
  for (int nt = 0; nt < 4; ++nt) {
    v8f acc = v8f_splat(b_fc[nt * 16 + (lane & 15)]);
#pragma unroll
    for (int kt = 0; kt < 8; ++kt)
      acc = wmma_f16(afr[kt], load_b(wfc, kt * 4 + nt, lane), acc);
    int c = nt * 16 + (lane & 15);
    int lane2 = lh * 16 + (c & 15);
#pragma unroll
    for (int j = 0; j < 8; ++j) {
      int m = j + 8 * (lane >> 4);
      int t = t0 + m;
      _Float16* base = opk + (size_t)(b * T + t) * 4096;
      base[((kt2 * 4 + nt) * 32 + lane2) * 16 + e] = (_Float16)acc[j];
    }
  }
}

// ---------------- K6: xf = PReLU(BN(x @ w_fpw^T)); write q (scaled 1/cf) + k pack ---

__global__ void __launch_bounds__(256) xf_kernel(
    const _Float16* __restrict__ xraw, const _Float16* __restrict__ wfp,
    const float* __restrict__ g1, const float* __restrict__ bb1,
    const float* __restrict__ m1, const float* __restrict__ v1,
    const float* __restrict__ pr1,
    _Float16* __restrict__ xfq, _Float16* __restrict__ xfk) {
  int wave = threadIdx.x >> 5, lane = threadIdx.x & 31;
  int mt = blockIdx.x * 8 + wave;           // rows r = (b*T+t)*64 + kbin
  int r0 = mt * 16;
  const _Float16* abase = xraw + (size_t)r0 * 128;
  v16h afr[4];
#pragma unroll
  for (int kt = 0; kt < 4; ++kt) afr[kt] = load_a(abase + kt * 32, 128, lane);
  int bt = r0 >> 6;
  int i0 = r0 & 63;
  float alpha = pr1[0];
  for (int nt = 0; nt < 4; ++nt) {
    v8f acc = v8f_splat(0.f);
#pragma unroll
    for (int kt = 0; kt < 4; ++kt)
      acc = wmma_f16(afr[kt], load_b(wfp, kt * 4 + nt, lane), acc);
    int c = nt * 16 + (lane & 15);
    float sc = g1[c] * rsqrtf(v1[c] + LN_EPS);
    float sh = bb1[c] - m1[c] * sc;
    int ckt = c >> 5, clh = (c >> 4) & 1, ce = c & 15;
    _Float16* kbase = xfk + (size_t)bt * 4096;
#pragma unroll
    for (int j = 0; j < 8; ++j) {
      int m = j + 8 * (lane >> 4);
      int ibin = i0 + m;
      float y = acc[j] * sc + sh;
      y = (y >= 0.f) ? y : alpha * y;
      // q gets 1/cf = 1/64 folded in (cf = sqrt(K*N/2) = 64)
      xfq[((size_t)bt * 64 + ibin) * 64 + c] = (_Float16)(y * (1.f / 64.f));
      kbase[((ckt * 4 + (ibin >> 4)) * 32 + (clh * 16 + (ibin & 15))) * 16 + ce] =
          (_Float16)y;
    }
  }
}

// ---------------- K7: fused banded attention + output projection + residual --------

__global__ void __launch_bounds__(128, 1) attn_out_kernel(
    const _Float16* __restrict__ xfq, const _Float16* __restrict__ xfk,
    const _Float16* __restrict__ opk, const _Float16* __restrict__ wout,
    const float* __restrict__ xin,
    const float* __restrict__ g2, const float* __restrict__ bb2,
    const float* __restrict__ m2, const float* __restrict__ v2,
    const float* __restrict__ pr2, const int* __restrict__ chunk,
    float* __restrict__ out) {
  const int T = 512, N = 128;
  __shared__ float attf[64 * 64];
  __shared__ _Float16 att16[64 * 64];
  __shared__ _Float16 of16[64 * 64];
  int tid = threadIdx.x, wave = tid >> 5, lane = tid & 31;
  int bt = blockIdx.x;
  int b = bt >> 9, t = bt & 511;
  int cs = chunk[0];
  // 1) att = (q/cf) @ k^T  — wave w computes rows 16w..16w+15
  {
    const _Float16* qb = xfq + ((size_t)bt * 64 + wave * 16) * 64;
    v16h afr[2];
#pragma unroll
    for (int kt = 0; kt < 2; ++kt) afr[kt] = load_a(qb + kt * 32, 64, lane);
    const _Float16* kb = xfk + (size_t)bt * 4096;
    for (int nt = 0; nt < 4; ++nt) {
      v8f acc = v8f_splat(0.f);
#pragma unroll
      for (int kt = 0; kt < 2; ++kt)
        acc = wmma_f16(afr[kt], load_b(kb, kt * 4 + nt, lane), acc);
      int n = nt * 16 + (lane & 15);
#pragma unroll
      for (int j = 0; j < 8; ++j)
        attf[(wave * 16 + j + 8 * (lane >> 4)) * 64 + n] = acc[j];
    }
  }
  __syncthreads();
  // 2) banded softmax (|i-j| < chunk), one row per thread
  if (tid < 64) {
    int i = tid;
    int lo = i - cs + 1; if (lo < 0) lo = 0;
    int hi = i + cs - 1; if (hi > 63) hi = 63;
    float mx = -1e30f;
    for (int j = lo; j <= hi; ++j) mx = fmaxf(mx, attf[i * 64 + j]);
    float sum = 0.f;
    for (int j = lo; j <= hi; ++j) sum += __expf(attf[i * 64 + j] - mx);
    float inv = 1.f / sum;
    for (int j = 0; j < 64; ++j) {
      float p = (j >= lo && j <= hi) ? __expf(attf[i * 64 + j] - mx) * inv : 0.f;
      att16[i * 64 + j] = (_Float16)p;
    }
  }
  __syncthreads();
  // 3) out_f = att @ out_t
  {
    v16h afr[2];
#pragma unroll
    for (int kt = 0; kt < 2; ++kt)
      afr[kt] = load_a(&att16[(wave * 16) * 64 + kt * 32], 64, lane);
    const _Float16* ob = opk + (size_t)bt * 4096;
    for (int nt = 0; nt < 4; ++nt) {
      v8f acc = v8f_splat(0.f);
#pragma unroll
      for (int kt = 0; kt < 2; ++kt)
        acc = wmma_f16(afr[kt], load_b(ob, kt * 4 + nt, lane), acc);
      int n = nt * 16 + (lane & 15);
#pragma unroll
      for (int j = 0; j < 8; ++j)
        of16[(wave * 16 + j + 8 * (lane >> 4)) * 64 + n] = (_Float16)acc[j];
    }
  }
  __syncthreads();
  // 4) y = PReLU(BN(out_f @ w_out^T)); out = inputs + y
  {
    float alpha = pr2[0];
    v16h afr[2];
#pragma unroll
    for (int kt = 0; kt < 2; ++kt)
      afr[kt] = load_a(&of16[(wave * 16) * 64 + kt * 32], 64, lane);
    for (int nt = 0; nt < 8; ++nt) {
      v8f acc = v8f_splat(0.f);
#pragma unroll
      for (int kt = 0; kt < 2; ++kt)
        acc = wmma_f16(afr[kt], load_b(wout, kt * 8 + nt, lane), acc);
      int n = nt * 16 + (lane & 15);
      float sc = g2[n] * rsqrtf(v2[n] + LN_EPS);
      float sh = bb2[n] - m2[n] * sc;
#pragma unroll
      for (int j = 0; j < 8; ++j) {
        int i = wave * 16 + j + 8 * (lane >> 4);
        float y = acc[j] * sc + sh;
        y = (y >= 0.f) ? y : alpha * y;
        size_t off = ((size_t)(b * N + n) * T + t) * 64 + i;
        out[off] = xin[off] + y;
      }
    }
  }
}

// ---------------- host-side launch -------------------------------------------------

extern "C" void kernel_launch(void* const* d_in, const int* in_sizes, int n_in,
                              void* d_out, int out_size, void* d_ws, size_t ws_size,
                              hipStream_t stream) {
  (void)in_sizes; (void)n_in; (void)out_size; (void)ws_size;
  const float* x     = (const float*)d_in[0];
  const float* ln_w  = (const float*)d_in[1];
  const float* ln_b  = (const float*)d_in[2];
  const float* w_ih  = (const float*)d_in[3];
  const float* w_hh  = (const float*)d_in[4];
  const float* b_ih  = (const float*)d_in[5];
  const float* b_hh  = (const float*)d_in[6];
  const float* w_fc  = (const float*)d_in[7];
  const float* b_fc  = (const float*)d_in[8];
  const float* w_fpw = (const float*)d_in[9];
  const float* bn1_g = (const float*)d_in[10];
  const float* bn1_b = (const float*)d_in[11];
  const float* bn1_m = (const float*)d_in[12];
  const float* bn1_v = (const float*)d_in[13];
  const float* pr1   = (const float*)d_in[14];
  const float* w_out = (const float*)d_in[15];
  const float* bn2_g = (const float*)d_in[16];
  const float* bn2_b = (const float*)d_in[17];
  const float* bn2_m = (const float*)d_in[18];
  const float* bn2_v = (const float*)d_in[19];
  const float* pr2   = (const float*)d_in[20];
  const int*   csz   = (const int*)d_in[21];
  float* out = (float*)d_out;

  // workspace carve (all in f16 elements; every offset 16-element aligned)
  _Float16* ws = (_Float16*)d_ws;
  size_t off = 0;
  _Float16* pk_wih  = ws + off; off += (size_t)768 * 128;
  _Float16* pk_whh  = ws + off; off += (size_t)768 * 256;
  _Float16* pk_wfc  = ws + off; off += (size_t)64 * 256;
  _Float16* pk_wfpw = ws + off; off += (size_t)64 * 128;
  _Float16* pk_wout = ws + off; off += (size_t)128 * 64;
  _Float16* xraw    = ws + off; off += (size_t)8 * 512 * 64 * 128;
  _Float16* xln     = ws + off; off += (size_t)8 * 512 * 64 * 128;
  _Float16* xp      = ws + off; off += (size_t)512 * 512 * 768;
  _Float16* hsb     = ws + off; off += (size_t)512 * 512 * 256;
  _Float16* opk     = ws + off; off += (size_t)4096 * 4096;
  _Float16* xfq     = ws + off; off += (size_t)4096 * 4096;
  _Float16* xfk     = ws + off; off += (size_t)4096 * 4096;

  // K1: pack all five weight matrices as WMMA B-fragments of W^T
  pack_b_kernel<<<(4 * 48 * 32 + 255) / 256, 256, 0, stream>>>(w_ih,  pk_wih,  768, 128);
  pack_b_kernel<<<(8 * 48 * 32 + 255) / 256, 256, 0, stream>>>(w_hh,  pk_whh,  768, 256);
  pack_b_kernel<<<(8 * 4 * 32 + 255) / 256, 256, 0, stream>>>(w_fc,  pk_wfc,  64, 256);
  pack_b_kernel<<<(4 * 4 * 32 + 255) / 256, 256, 0, stream>>>(w_fpw, pk_wfpw, 64, 128);
  pack_b_kernel<<<(2 * 8 * 32 + 255) / 256, 256, 0, stream>>>(w_out, pk_wout, 128, 64);

  // K2: transpose + LayerNorm + f16 convert
  ln_convert_kernel<<<8 * 512, 256, 0, stream>>>(x, ln_w, ln_b, xraw, xln);

  // K3: xp = LN(x) @ w_ih^T + b_ih
  gemm_xp_kernel<<<2048, 256, 0, stream>>>(xln, pk_wih, b_ih, xp);

  // K4: GRU recurrence (32 persistent workgroups x 16 sequences, TDM-staged xp)
  gru_kernel<<<32, 256, 0, stream>>>(xp, pk_whh, b_hh, hsb);

  // K5: out_t = hs @ w_fc^T + b_fc -> packed B fragments per (b,t)
  gemm_outt_kernel<<<2048, 256, 0, stream>>>(hsb, pk_wfc, b_fc, opk);

  // K6: freq branch conv + BN + PReLU -> q (scaled) and k-pack
  xf_kernel<<<2048, 256, 0, stream>>>(xraw, pk_wfpw, bn1_g, bn1_b, bn1_m, bn1_v,
                                      pr1, xfq, xfk);

  // K7: fused banded attention + w_out + BN + PReLU + residual
  attn_out_kernel<<<8 * 512, 128, 0, stream>>>(xfq, xfk, opk, pk_wout, x,
                                               bn2_g, bn2_b, bn2_m, bn2_v, pr2,
                                               csz, out);
}
